// GAT_80092550136008
// MI455X (gfx1250) — compile-verified, matrix operand order checked by reading, not run on hardware
//
#include <hip/hip_runtime.h>
#include <hip/hip_bf16.h>
#include <math.h>

typedef _Float16 half_t;
typedef __attribute__((ext_vector_type(16))) _Float16 v16h;
typedef __attribute__((ext_vector_type(8)))  _Float16 v8h;
typedef __attribute__((ext_vector_type(8)))  float    v8f;

#define NODES 50000
#define EDGES 800000
#define F_IN  165
#define HID   32
#define H1    8
#define H3    4
#define D1    256
#define D3    128
#define KPADX 192
#define BN_EPS 1e-5f

// ---------------------------------------------------------------------------
// WMMA f16 GEMM: C[M,Nc] = A[M,K] (f16, row-major) x Bt[Nc,K]^T (f16, pre-transposed)
// One wave per block; each wave owns a 16x64 output tile (4 WMMA accumulators).
// grid.x = M/16, grid.y = Nc/64.  K multiple of 32, M multiple of 16.
// ---------------------------------------------------------------------------
__device__ inline v16h load_frag16(const half_t* __restrict__ p) {
    // Two contiguous 16-byte segments: halves [0..7] at p, [8..15] at p+16
    v8h lo = *(const v8h*)(p);
    v8h hi = *(const v8h*)(p + 16);
    v16h r;
#pragma unroll
    for (int i = 0; i < 8; ++i) { r[i] = lo[i]; r[i + 8] = hi[i]; }
    return r;
}

__global__ __launch_bounds__(32)
void gemm_wmma_f16(const half_t* __restrict__ A, const half_t* __restrict__ Bt,
                   float* __restrict__ C, int M, int K, int Nc) {
    const int lane = threadIdx.x;
    const int l15  = lane & 15;
    const int base = (lane < 16) ? 0 : 8;          // K sub-offset per half-wave
    const int row  = blockIdx.x * 16 + l15;         // A row for this lane
    const int n0   = blockIdx.y * 64;

    const half_t* __restrict__ arow = A  + (size_t)row * K;
    const half_t* __restrict__ bc0  = Bt + (size_t)(n0 +  0 + l15) * K;
    const half_t* __restrict__ bc1  = Bt + (size_t)(n0 + 16 + l15) * K;
    const half_t* __restrict__ bc2  = Bt + (size_t)(n0 + 32 + l15) * K;
    const half_t* __restrict__ bc3  = Bt + (size_t)(n0 + 48 + l15) * K;

    v8f c0 = {}, c1 = {}, c2 = {}, c3 = {};

    for (int k0 = 0; k0 < K; k0 += 32) {
        v16h a  = load_frag16(arow + k0 + base);
        v16h b0 = load_frag16(bc0  + k0 + base);
        v16h b1 = load_frag16(bc1  + k0 + base);
        v16h b2 = load_frag16(bc2  + k0 + base);
        v16h b3 = load_frag16(bc3  + k0 + base);
        c0 = __builtin_amdgcn_wmma_f32_16x16x32_f16(false, a, false, b0, (short)0, c0, false, false);
        c1 = __builtin_amdgcn_wmma_f32_16x16x32_f16(false, a, false, b1, (short)0, c1, false, false);
        c2 = __builtin_amdgcn_wmma_f32_16x16x32_f16(false, a, false, b2, (short)0, c2, false, false);
        c3 = __builtin_amdgcn_wmma_f32_16x16x32_f16(false, a, false, b3, (short)0, c3, false, false);
    }

    // C/D layout: VGPR r -> row = r + 8*(lane/16); col = lane&15 within 16-wide tile
    const int mrow0 = blockIdx.x * 16 + ((lane < 16) ? 0 : 8);
#pragma unroll
    for (int r = 0; r < 8; ++r) {
        size_t rowoff = (size_t)(mrow0 + r) * Nc + n0 + l15;
        C[rowoff +  0] = c0[r];
        C[rowoff + 16] = c1[r];
        C[rowoff + 32] = c2[r];
        C[rowoff + 48] = c3[r];
    }
}

// ---------------------------------------------------------------------------
// Conversion / init kernels
// ---------------------------------------------------------------------------
__global__ void cast_pad_f16(const float* __restrict__ src, half_t* __restrict__ dst,
                             int M, int Ksrc, int Kdst, float scale) {
    int idx = blockIdx.x * blockDim.x + threadIdx.x;
    if (idx >= M * Kdst) return;
    int r = idx / Kdst, k = idx - r * Kdst;
    dst[idx] = (k < Ksrc) ? (half_t)(src[(size_t)r * Ksrc + k] * scale) : (half_t)0.f;
}

// W[K,Dout] (row-major f32) -> Wt[Dout,Kpad] f16 (transposed, zero-padded)
__global__ void castW_t(const float* __restrict__ W, half_t* __restrict__ Wt,
                        int K, int Dout, int Kpad) {
    int idx = blockIdx.x * blockDim.x + threadIdx.x;
    if (idx >= Dout * Kpad) return;
    int o = idx / Kpad, k = idx - o * Kpad;
    Wt[idx] = (k < K) ? (half_t)W[(size_t)k * Dout + o] : (half_t)0.f;
}

__global__ void fill_f32(float* __restrict__ p, float v, int n) {
    int i = blockIdx.x * blockDim.x + threadIdx.x;
    if (i < n) p[i] = v;
}

// ---------------------------------------------------------------------------
// Attention coefficients: as/ad[n,h] = sum_d h[n,h*32+d] * a_{s,d}[h,d]
// ---------------------------------------------------------------------------
__global__ void att_coef(const float* __restrict__ h, const float* __restrict__ a_s,
                         const float* __restrict__ a_d, float* __restrict__ as_o,
                         float* __restrict__ ad_o, int n, int heads) {
    int idx = blockIdx.x * blockDim.x + threadIdx.x;
    if (idx >= n * heads) return;
    int nn = idx / heads, hh = idx - nn * heads;
    const float* hp = h + (size_t)nn * heads * HID + hh * HID;
    float ss = 0.f, dd = 0.f;
#pragma unroll
    for (int d = 0; d < HID; ++d) {
        float v = hp[d];
        ss += v * a_s[hh * HID + d];
        dd += v * a_d[hh * HID + d];
    }
    as_o[idx] = ss; ad_o[idx] = dd;
}

// ---------------------------------------------------------------------------
// Segment-softmax passes over edges
// ---------------------------------------------------------------------------
__device__ inline void atomicMaxF(float* addr, float val) {
    if (val >= 0.f) atomicMax((int*)addr, __float_as_int(val));
    else            atomicMin((unsigned int*)addr, __float_as_uint(val));
}
__device__ inline float lrelu(float v) { return v > 0.f ? v : 0.2f * v; }

__global__ void edge_max(const int* __restrict__ ei, const float* __restrict__ as_,
                         const float* __restrict__ ad_, float* __restrict__ mbuf,
                         int E, int heads) {
    int idx = blockIdx.x * blockDim.x + threadIdx.x;
    if (idx >= E * heads) return;
    int e = idx / heads, hh = idx - e * heads;
    int s = ei[e], d = ei[E + e];
    float v = lrelu(as_[s * heads + hh] + ad_[d * heads + hh]);
    atomicMaxF(&mbuf[d * heads + hh], v);
}

__global__ void self_max(const float* __restrict__ as_, const float* __restrict__ ad_,
                         float* __restrict__ mbuf, int n, int heads) {
    int idx = blockIdx.x * blockDim.x + threadIdx.x;
    if (idx >= n * heads) return;
    float v = lrelu(as_[idx] + ad_[idx]);
    mbuf[idx] = fmaxf(mbuf[idx], v);
}

__global__ void edge_exp(const int* __restrict__ ei, const float* __restrict__ as_,
                         const float* __restrict__ ad_, const float* __restrict__ mbuf,
                         float* __restrict__ exbuf, float* __restrict__ den,
                         int E, int heads) {
    int idx = blockIdx.x * blockDim.x + threadIdx.x;
    if (idx >= E * heads) return;
    int e = idx / heads, hh = idx - e * heads;
    int s = ei[e], d = ei[E + e];
    float v  = lrelu(as_[s * heads + hh] + ad_[d * heads + hh]);
    float ex = expf(v - mbuf[d * heads + hh]);
    exbuf[idx] = ex;
    atomicAdd(&den[d * heads + hh], ex);
}

// 32 lanes per (edge,head): lane = channel within head. Coalesced gather+scatter.
__global__ __launch_bounds__(256)
void edge_scatter(const int* __restrict__ ei, const float* __restrict__ exbuf,
                  const float* __restrict__ hlin, float* __restrict__ agg,
                  int E, int heads) {
    int lane = threadIdx.x & 31;
    int grp  = (blockIdx.x * blockDim.x + threadIdx.x) >> 5;
    if (grp >= E * heads) return;
    int e = grp / heads, hh = grp - e * heads;
    int s = ei[e], d = ei[E + e];
    float ex = exbuf[grp];
    int D = heads * HID;
    float v = hlin[(size_t)s * D + hh * HID + lane];
    atomicAdd(&agg[(size_t)d * D + hh * HID + lane], ex * v);
}

// ---------------------------------------------------------------------------
// Per-node epilogues
// ---------------------------------------------------------------------------
__global__ void epilogue1(const float* __restrict__ agg, const float* __restrict__ den,
                          const float* __restrict__ b, const float* __restrict__ g,
                          const float* __restrict__ be, const float* __restrict__ mean,
                          const float* __restrict__ var,
                          float* __restrict__ act_f32, half_t* __restrict__ act_f16,
                          int n, int D, int heads) {
    int idx = blockIdx.x * blockDim.x + threadIdx.x;
    if (idx >= n * D) return;
    int nn = idx / D, c = idx - nn * D, hh = c / HID;
    float val = agg[idx] / (den[nn * heads + hh] + 1e-16f) + b[c];
    val = (val - mean[c]) * rsqrtf(var[c] + BN_EPS) * g[c] + be[c];
    val = val > 0.f ? val : expm1f(val);          // ELU
    act_f32[idx] = val;
    act_f16[idx] = (half_t)val;
}

__global__ void epilogue2(const float* __restrict__ agg, const float* __restrict__ den,
                          const float* __restrict__ mbuf, const float* __restrict__ as_,
                          const float* __restrict__ ad_, const float* __restrict__ hlin,
                          const float* __restrict__ b, const float* __restrict__ g,
                          const float* __restrict__ be, const float* __restrict__ mean,
                          const float* __restrict__ var, const float* __restrict__ resid,
                          half_t* __restrict__ act_f16, const float* __restrict__ temp,
                          int n, int D, int heads) {
    int idx = blockIdx.x * blockDim.x + threadIdx.x;
    if (idx >= n * D) return;
    int nn = idx / D, c = idx - nn * D, hh = c / HID;
    int nh = nn * heads + hh;
    float sv = lrelu(as_[nh] + ad_[nh]);
    float es = expf(sv - mbuf[nh]);
    float dt = den[nh] + es;
    float val = (agg[idx] + es * hlin[idx]) / (dt + 1e-16f) + b[c];
    val = (val - mean[c]) * rsqrtf(var[c] + BN_EPS) * g[c] + be[c];
    val += resid[idx];
    val = val > 0.f ? val : expm1f(val);          // ELU
    act_f16[idx] = (half_t)(val * temp[0]);       // fold 'temp' into layer-3 input
}

__global__ void epilogue3(const float* __restrict__ agg, const float* __restrict__ den,
                          const float* __restrict__ mbuf, const float* __restrict__ as_,
                          const float* __restrict__ ad_, const float* __restrict__ hlin,
                          const float* __restrict__ b, const float* __restrict__ xinit,
                          const float* __restrict__ skb, float* __restrict__ h3,
                          int n, int D, int heads) {
    int idx = blockIdx.x * blockDim.x + threadIdx.x;
    if (idx >= n * D) return;
    int nn = idx / D, c = idx - nn * D, hh = c / HID;
    int nh = nn * heads + hh;
    float sv = lrelu(as_[nh] + ad_[nh]);
    float es = expf(sv - mbuf[nh]);
    float dt = den[nh] + es;
    float val = (agg[idx] + es * hlin[idx]) / (dt + 1e-16f) + b[c];
    h3[idx] = val + xinit[idx] + skb[c];
}

__global__ void fc_logsoftmax(const float* __restrict__ h3, const float* __restrict__ fcW,
                              const float* __restrict__ fcb, float* __restrict__ out, int n) {
    int nn = blockIdx.x * blockDim.x + threadIdx.x;
    if (nn >= n) return;
    const float* hp = h3 + (size_t)nn * D3;
    float z0 = fcb[0], z1 = fcb[1];
#pragma unroll 4
    for (int c = 0; c < D3; ++c) {
        float v = hp[c];
        z0 += v * fcW[c * 2 + 0];
        z1 += v * fcW[c * 2 + 1];
    }
    float mx = fmaxf(z0, z1);
    float l  = mx + logf(expf(z0 - mx) + expf(z1 - mx));
    out[nn * 2 + 0] = z0 - l;
    out[nn * 2 + 1] = z1 - l;
}

// ---------------------------------------------------------------------------
// Host orchestration
// ---------------------------------------------------------------------------
static inline dim3 g1d(long total, int bs) { return dim3((unsigned)((total + bs - 1) / bs)); }

extern "C" void kernel_launch(void* const* d_in, const int* in_sizes, int n_in,
                              void* d_out, int out_size, void* d_ws, size_t ws_size,
                              hipStream_t stream) {
    const float* x    = (const float*)d_in[0];
    const int*   ei   = (const int*)  d_in[1];
    const float* W1   = (const float*)d_in[2];
    const float* a1s  = (const float*)d_in[3];
    const float* a1d  = (const float*)d_in[4];
    const float* b1   = (const float*)d_in[5];
    const float* g1   = (const float*)d_in[6];
    const float* be1  = (const float*)d_in[7];
    const float* m1   = (const float*)d_in[8];
    const float* v1   = (const float*)d_in[9];
    const float* W2   = (const float*)d_in[10];
    const float* a2s  = (const float*)d_in[11];
    const float* a2d  = (const float*)d_in[12];
    const float* b2   = (const float*)d_in[13];
    const float* g2   = (const float*)d_in[14];
    const float* be2  = (const float*)d_in[15];
    const float* m2   = (const float*)d_in[16];
    const float* v2   = (const float*)d_in[17];
    const float* W3   = (const float*)d_in[18];
    const float* a3s  = (const float*)d_in[19];
    const float* a3d  = (const float*)d_in[20];
    const float* b3   = (const float*)d_in[21];
    const float* fcW  = (const float*)d_in[22];
    const float* fcb  = (const float*)d_in[23];
    const float* skW  = (const float*)d_in[24];
    const float* skb  = (const float*)d_in[25];
    const float* temp = (const float*)d_in[26];
    float* out = (float*)d_out;

    // Workspace carve
    char* base = (char*)d_ws;
    size_t off = 0;
    auto alloc = [&](size_t bytes) -> void* {
        off = (off + 255) & ~(size_t)255;
        void* p = base + off; off += bytes; return p;
    };
    half_t* a16   = (half_t*)alloc((size_t)NODES * D1 * 2);   // f16 A (x padded 192 / acts 256)
    half_t* wt16  = (half_t*)alloc((size_t)D1 * D1 * 2);      // f16 transposed weights
    half_t* skwt  = (half_t*)alloc((size_t)D3 * KPADX * 2);
    float*  hlin  = (float*)alloc((size_t)NODES * D1 * 4);    // linear transform output
    float*  agg   = (float*)alloc((size_t)NODES * D1 * 4);    // attention-weighted sum
    float*  act1  = (float*)alloc((size_t)NODES * D1 * 4);    // layer-1 activation (residual)
    float*  asb   = (float*)alloc((size_t)NODES * H1 * 4);
    float*  adb   = (float*)alloc((size_t)NODES * H1 * 4);
    float*  mbuf  = (float*)alloc((size_t)NODES * H1 * 4);
    float*  den   = (float*)alloc((size_t)NODES * H1 * 4);
    float*  exbuf = (float*)alloc((size_t)EDGES * H1 * 4);
    float*  xinit = (float*)alloc((size_t)NODES * D3 * 4);
    float*  h3    = (float*)alloc((size_t)NODES * D3 * 4);
    (void)ws_size; (void)n_in; (void)in_sizes; (void)out_size;

    const int BS = 256;
    const float NEG_INF = -__builtin_huge_valf();

    // ---- x_init = x @ skW (+skb later) ----
    cast_pad_f16<<<g1d((long)NODES * KPADX, BS), BS, 0, stream>>>(x, a16, NODES, F_IN, KPADX, 1.f);
    castW_t<<<g1d((long)D3 * KPADX, BS), BS, 0, stream>>>(skW, skwt, F_IN, D3, KPADX);
    gemm_wmma_f16<<<dim3(NODES / 16, D3 / 64), 32, 0, stream>>>(a16, skwt, xinit, NODES, KPADX, D3);

    // ---- Layer 1: GAT(165 -> 8x32), no self loops ----
    castW_t<<<g1d((long)D1 * KPADX, BS), BS, 0, stream>>>(W1, wt16, F_IN, D1, KPADX);
    gemm_wmma_f16<<<dim3(NODES / 16, D1 / 64), 32, 0, stream>>>(a16, wt16, hlin, NODES, KPADX, D1);
    att_coef<<<g1d((long)NODES * H1, BS), BS, 0, stream>>>(hlin, a1s, a1d, asb, adb, NODES, H1);
    fill_f32<<<g1d((long)NODES * H1, BS), BS, 0, stream>>>(mbuf, NEG_INF, NODES * H1);
    fill_f32<<<g1d((long)NODES * H1, BS), BS, 0, stream>>>(den, 0.f, NODES * H1);
    fill_f32<<<g1d((long)NODES * D1, BS), BS, 0, stream>>>(agg, 0.f, NODES * D1);
    edge_max<<<g1d((long)EDGES * H1, BS), BS, 0, stream>>>(ei, asb, adb, mbuf, EDGES, H1);
    edge_exp<<<g1d((long)EDGES * H1, BS), BS, 0, stream>>>(ei, asb, adb, mbuf, exbuf, den, EDGES, H1);
    edge_scatter<<<g1d((long)EDGES * H1 * 32, BS), BS, 0, stream>>>(ei, exbuf, hlin, agg, EDGES, H1);
    epilogue1<<<g1d((long)NODES * D1, BS), BS, 0, stream>>>(agg, den, b1, g1, be1, m1, v1,
                                                           act1, a16, NODES, D1, H1);

    // ---- Layer 2: GAT(256 -> 8x32), self loops, residual ----
    castW_t<<<g1d((long)D1 * D1, BS), BS, 0, stream>>>(W2, wt16, D1, D1, D1);
    gemm_wmma_f16<<<dim3(NODES / 16, D1 / 64), 32, 0, stream>>>(a16, wt16, hlin, NODES, D1, D1);
    att_coef<<<g1d((long)NODES * H1, BS), BS, 0, stream>>>(hlin, a2s, a2d, asb, adb, NODES, H1);
    fill_f32<<<g1d((long)NODES * H1, BS), BS, 0, stream>>>(mbuf, NEG_INF, NODES * H1);
    fill_f32<<<g1d((long)NODES * H1, BS), BS, 0, stream>>>(den, 0.f, NODES * H1);
    fill_f32<<<g1d((long)NODES * D1, BS), BS, 0, stream>>>(agg, 0.f, NODES * D1);
    edge_max<<<g1d((long)EDGES * H1, BS), BS, 0, stream>>>(ei, asb, adb, mbuf, EDGES, H1);
    self_max<<<g1d((long)NODES * H1, BS), BS, 0, stream>>>(asb, adb, mbuf, NODES, H1);
    edge_exp<<<g1d((long)EDGES * H1, BS), BS, 0, stream>>>(ei, asb, adb, mbuf, exbuf, den, EDGES, H1);
    edge_scatter<<<g1d((long)EDGES * H1 * 32, BS), BS, 0, stream>>>(ei, exbuf, hlin, agg, EDGES, H1);
    epilogue2<<<g1d((long)NODES * D1, BS), BS, 0, stream>>>(agg, den, mbuf, asb, adb, hlin,
                                                           b2, g2, be2, m2, v2, act1,
                                                           a16, temp, NODES, D1, H1);

    // ---- Layer 3: GAT(256 -> 4x32), self loops ----
    castW_t<<<g1d((long)D3 * D1, BS), BS, 0, stream>>>(W3, wt16, D1, D3, D1);
    gemm_wmma_f16<<<dim3(NODES / 16, D3 / 64), 32, 0, stream>>>(a16, wt16, hlin, NODES, D1, D3);
    att_coef<<<g1d((long)NODES * H3, BS), BS, 0, stream>>>(hlin, a3s, a3d, asb, adb, NODES, H3);
    fill_f32<<<g1d((long)NODES * H3, BS), BS, 0, stream>>>(mbuf, NEG_INF, NODES * H3);
    fill_f32<<<g1d((long)NODES * H3, BS), BS, 0, stream>>>(den, 0.f, NODES * H3);
    fill_f32<<<g1d((long)NODES * D3, BS), BS, 0, stream>>>(agg, 0.f, NODES * D3);
    edge_max<<<g1d((long)EDGES * H3, BS), BS, 0, stream>>>(ei, asb, adb, mbuf, EDGES, H3);
    self_max<<<g1d((long)NODES * H3, BS), BS, 0, stream>>>(asb, adb, mbuf, NODES, H3);
    edge_exp<<<g1d((long)EDGES * H3, BS), BS, 0, stream>>>(ei, asb, adb, mbuf, exbuf, den, EDGES, H3);
    edge_scatter<<<g1d((long)EDGES * H3 * 32, BS), BS, 0, stream>>>(ei, exbuf, hlin, agg, EDGES, H3);
    epilogue3<<<g1d((long)NODES * D3, BS), BS, 0, stream>>>(agg, den, mbuf, asb, adb, hlin,
                                                           b3, xinit, skb, h3, NODES, D3, H3);

    // ---- FC + log_softmax ----
    fc_logsoftmax<<<g1d((long)NODES, BS), BS, 0, stream>>>(h3, fcW, fcb, out, NODES);
}
